// GPRGNN_21801253994544
// MI455X (gfx1250) — matrix operand order repaired and checked by reading, not compiled, not run on hardware
//
#include <hip/hip_runtime.h>
#include <hip/hip_bf16.h>

#define NFEAT  512
#define HIDDEN 256
#define NCLASS 64
#define KPROP  10

typedef __attribute__((ext_vector_type(16))) __bf16 v16bf;
typedef __attribute__((ext_vector_type(8)))  float  v8f;

#if defined(__has_builtin)
#if __has_builtin(__builtin_amdgcn_tensor_load_to_lds) && __has_builtin(__builtin_amdgcn_s_wait_tensorcnt)
#define HAVE_TDM 1
#endif
#endif

#ifdef HAVE_TDM
typedef unsigned int v4u __attribute__((ext_vector_type(4)));
typedef int          v8i __attribute__((ext_vector_type(8)));
typedef int          v4i __attribute__((ext_vector_type(4)));
#endif

// ---- WMMA lane/element maps (CDNA5 ISA 7.12.2, wave32) ----
// A 16x32 bf16: lane = hi*16 + m (m = row). element j -> K:
//   lo lanes: K = j (j<8), 16+(j-8) (j>=8); hi lanes: +8
// B 32x16 bf16: lane = hi*16 + n (n = col). element j -> K as A; code uses K = j + 16*hi for A and B
// C/D 16x16 f32: lane = hi*16 + n.  element j -> M = j + 8*hi

// ------------------------------------------------------------------
// gcn_norm helpers
// ------------------------------------------------------------------
__global__ void k_init_deg(float* deg, int n) {
  int i = blockIdx.x * blockDim.x + threadIdx.x;
  if (i < n) deg[i] = 1.0f;  // self-loop
}

__global__ void k_edge_deg(const long long* __restrict__ dst, float* deg, int e) {
  int i = blockIdx.x * blockDim.x + threadIdx.x;
  if (i < e) unsafeAtomicAdd(&deg[(int)dst[i]], 1.0f);
}

__global__ void k_rsqrt(float* d, int n) {
  int i = blockIdx.x * blockDim.x + threadIdx.x;
  if (i < n) {
    float v = d[i];
    d[i] = v > 0.f ? rsqrtf(v) : 0.f;
  }
}

__global__ void k_norm(const long long* __restrict__ src,
                       const long long* __restrict__ dst,
                       const float* __restrict__ dinv,
                       float* __restrict__ nrm, int e) {
  int i = blockIdx.x * blockDim.x + threadIdx.x;
  if (i < e) nrm[i] = dinv[(int)src[i]] * dinv[(int)dst[i]];
}

// ------------------------------------------------------------------
// One-shot weight transpose + fp32->bf16:  w[K][Ncol] -> wt[Ncol][K]
// (writes contiguous in k; tiny: 128K + 16K elements total)
// ------------------------------------------------------------------
__global__ void k_cvt_wT(const float* __restrict__ w, __bf16* __restrict__ wt,
                         int K, int Ncol) {
  int t = blockIdx.x * blockDim.x + threadIdx.x;
  if (t < K * Ncol) {
    int k = t % K;
    int n = t / K;
    wt[(size_t)n * K + k] = (__bf16)w[(size_t)k * Ncol + n];
  }
}

// ------------------------------------------------------------------
// Fused MLP:  h = relu(X@W1+b1)@W2+b2   (bf16 WMMA, fp32 accumulate)
// 256 threads = 8 waves; 32 rows of X per workgroup.
// X tile (32x512 fp32, 64 KB) staged in LDS via Tensor Data Mover.
// B fragments: contiguous 32 B v16bf loads from pre-transposed bf16 weights.
// ------------------------------------------------------------------
__global__ __launch_bounds__(256) void k_mlp(const float* __restrict__ x,
                                             const __bf16* __restrict__ Wt1,
                                             const float* __restrict__ b1,
                                             const __bf16* __restrict__ Wt2,
                                             const float* __restrict__ b2,
                                             float* __restrict__ h) {
  __shared__ __align__(16) float  lds_x[32 * NFEAT];    // 64 KB
  __shared__ __align__(16) __bf16 lds_h1[32 * HIDDEN];  // 16 KB

  const int tid  = threadIdx.x;
  const int wave = tid >> 5;
  const int lane = tid & 31;
  const int m    = lane & 15;  // row (A) / col (B,C) within tile
  const int hi   = lane >> 4;
  const int rowBase = blockIdx.x * 32;

  // ---------- Stage X block into LDS ----------
#ifdef HAVE_TDM
  if (tid < 32) {
    // D# group0 (ISA 8.3): count=1 | lds_addr | global_addr(57b) | type=2
    unsigned long long ga = (unsigned long long)(uintptr_t)(x + (size_t)rowBase * NFEAT);
    unsigned int ldsa = (unsigned int)(uintptr_t)(&lds_x[0]);
    v4u g0;
    g0[0] = 1u;                                   // count=1 (valid user descriptor)
    g0[1] = ldsa;                                 // lds_addr, bits 63:32
    g0[2] = (unsigned int)(ga & 0xFFFFFFFFu);     // global_addr[31:0]
    g0[3] = (unsigned int)((ga >> 32) & 0x1FFFFFFu) | (2u << 30);  // addr[56:32] | type=2
    // D# group1 (ISA 8.4): data_size=2 (4B); tensor 512x32; tile 512x32; stride 512
    v8i g1;
    g1[0] = (int)(2u << 16);                           // data_size = 4 bytes
    g1[1] = (int)(((unsigned)NFEAT & 0xFFFFu) << 16);  // tensor_dim0 lo16 (bits 63:48)
    g1[2] = (int)((((unsigned)NFEAT >> 16) & 0xFFFFu)  // tensor_dim0 hi16
                  | ((32u & 0xFFFFu) << 16));          // tensor_dim1 lo16
    g1[3] = (int)(((unsigned)NFEAT & 0xFFFFu) << 16);  // tensor_dim1 hi16=0 | tile_dim0=512
    g1[4] = (int)32;                                   // tile_dim1=32, tile_dim2=0
    g1[5] = (int)NFEAT;                                // tensor_dim0_stride lo32
    g1[6] = 0;                                         // stride hi | dim1_stride lo
    g1[7] = 0;
    v4i gz4 = {0, 0, 0, 0};
    v8i gz8 = {0, 0, 0, 0, 0, 0, 0, 0};
    __builtin_amdgcn_tensor_load_to_lds(g0, g1, gz4, gz4, gz8, 0);
    __builtin_amdgcn_s_wait_tensorcnt((short)0);
  }
#else
  {
    const float4* s4 = (const float4*)(x + (size_t)rowBase * NFEAT);
    float4* d4 = (float4*)lds_x;
#pragma unroll
    for (int i = 0; i < (32 * NFEAT / 4) / 256; ++i)
      d4[tid + i * 256] = s4[tid + i * 256];
  }
#endif
  __syncthreads();

  // ---------- Phase 1: H1 = relu(X_blk @ W1 + b1) -> LDS (bf16) ----------
  {
    const int r  = wave >> 2;  // row tile 0..1
    const int cg = wave & 3;   // col tiles 4*cg .. 4*cg+3  (16 col tiles total)
    v8f acc[4] = {};
    for (int kt = 0; kt < NFEAT / 32; ++kt) {
      const int k0 = kt * 32;
      // A fragment: contiguous 16 floats from LDS at k0+16*hi, cvt to bf16
      const float* ax = lds_x + (16 * r + m) * NFEAT + k0 + 16 * hi;
      float4 p0 = *(const float4*)(ax);
      float4 p1 = *(const float4*)(ax + 4);
      float4 p2 = *(const float4*)(ax + 8);
      float4 p3 = *(const float4*)(ax + 12);
      v16bf a;
      a[0] = (__bf16)p0.x;  a[1] = (__bf16)p0.y;  a[2] = (__bf16)p0.z;  a[3] = (__bf16)p0.w;
      a[4] = (__bf16)p1.x;  a[5] = (__bf16)p1.y;  a[6] = (__bf16)p1.z;  a[7] = (__bf16)p1.w;
      a[8] = (__bf16)p2.x;  a[9] = (__bf16)p2.y;  a[10] = (__bf16)p2.z; a[11] = (__bf16)p2.w;
      a[12] = (__bf16)p3.x; a[13] = (__bf16)p3.y; a[14] = (__bf16)p3.z; a[15] = (__bf16)p3.w;
#pragma unroll
      for (int c4 = 0; c4 < 4; ++c4) {
        const int ct = 4 * cg + c4;
        // B fragment: contiguous 16 bf16 (32 B) from Wt1[n][k0+16*hi ..]
        v16bf b = *(const v16bf*)(Wt1 + (size_t)(ct * 16 + m) * NFEAT + k0 + 16 * hi);
        acc[c4] = __builtin_amdgcn_wmma_f32_16x16x32_bf16(
            false, a, false, b, (short)0, acc[c4], false, false);
      }
    }
#pragma unroll
    for (int c4 = 0; c4 < 4; ++c4) {
      const int n    = (4 * cg + c4) * 16 + m;
      const float bs = b1[n];
#pragma unroll
      for (int j = 0; j < 8; ++j) {
        float v = acc[c4][j] + bs;
        v = v > 0.f ? v : 0.f;  // relu
        lds_h1[(16 * r + j + 8 * hi) * HIDDEN + n] = (__bf16)v;
      }
    }
  }
  __syncthreads();

  // ---------- Phase 2: H = H1 @ W2 + b2 -> global (fp32) ----------
  {
    const int r = wave >> 2;  // row tile 0..1
    const int c = wave & 3;   // col tile 0..3
    v8f acc = {};
    for (int kt = 0; kt < HIDDEN / 32; ++kt) {
      const int k0 = kt * 32;
      // A fragment: contiguous 16 bf16 from LDS at k0+16*hi
      const __bf16* ap = lds_h1 + (16 * r + m) * HIDDEN + k0 + 16 * hi;
      v16bf a;
#pragma unroll
      for (int j = 0; j < 8; ++j) a[j] = ap[j];
#pragma unroll
      for (int j = 0; j < 8; ++j) a[8 + j] = ap[8 + j];
      // B fragment: contiguous 16 bf16 (32 B) from Wt2[n][k0+16*hi ..]
      v16bf b = *(const v16bf*)(Wt2 + (size_t)(c * 16 + m) * HIDDEN + k0 + 16 * hi);
      acc = __builtin_amdgcn_wmma_f32_16x16x32_bf16(
          false, a, false, b, (short)0, acc, false, false);
    }
    const int n    = c * 16 + m;
    const float bs = b2[n];
#pragma unroll
    for (int j = 0; j < 8; ++j)
      h[(size_t)(rowBase + 16 * r + j + 8 * hi) * NCLASS + n] = acc[j] + bs;
  }
}

// ------------------------------------------------------------------
// GPR propagation (h fits in 192 MB L2: gathers + atomics stay on-chip)
// ------------------------------------------------------------------
__global__ void k_scale_init(const float4* __restrict__ h,
                             const float* __restrict__ temp,
                             float4* __restrict__ out, int n4) {
  int i = blockIdx.x * blockDim.x + threadIdx.x;
  if (i < n4) {
    float t = temp[0];
    float4 v = h[i];
    out[i] = make_float4(t * v.x, t * v.y, t * v.z, t * v.w);
  }
}

// self-loop term: hn[i] = dinv[node]^2 * h[i]   (16 float4 per node row)
__global__ void k_self_init(const float4* __restrict__ h,
                            const float* __restrict__ dinv,
                            float4* __restrict__ hn, int n4) {
  int i = blockIdx.x * blockDim.x + threadIdx.x;
  if (i < n4) {
    float d = dinv[i >> 4];  // 64 floats = 16 float4 per node
    float s = d * d;
    float4 v = h[i];
    hn[i] = make_float4(s * v.x, s * v.y, s * v.z, s * v.w);
  }
}

// 16 threads per edge, 4 features each: hn[dst*64+f..f+3] += norm*h[src*64+f..f+3]
__global__ void k_edge_prop(const long long* __restrict__ src,
                            const long long* __restrict__ dst,
                            const float* __restrict__ nrm,
                            const float* __restrict__ h,
                            float* __restrict__ hn, int e) {
  long long t = (long long)blockIdx.x * blockDim.x + threadIdx.x;
  int ei = (int)(t >> 4);
  int f4 = (int)(t & 15) * 4;
  if (ei < e) {
    int r = (int)src[ei];
    int c = (int)dst[ei];
    float w = nrm[ei];
    float4 hv = *(const float4*)(h + (size_t)r * NCLASS + f4);
    float* op = hn + (size_t)c * NCLASS + f4;
    unsafeAtomicAdd(op + 0, w * hv.x);
    unsafeAtomicAdd(op + 1, w * hv.y);
    unsafeAtomicAdd(op + 2, w * hv.z);
    unsafeAtomicAdd(op + 3, w * hv.w);
  }
}

__global__ void k_accum(const float4* __restrict__ hn,
                        const float* __restrict__ tk,
                        float4* __restrict__ out, int n4) {
  int i = blockIdx.x * blockDim.x + threadIdx.x;
  if (i < n4) {
    float t = tk[0];
    float4 v = hn[i];
    float4 o = out[i];
    out[i] = make_float4(o.x + t * v.x, o.y + t * v.y, o.z + t * v.z, o.w + t * v.w);
  }
}

// ------------------------------------------------------------------
// In-place log_softmax over 64 classes: one wave32 per row, 2 vals/lane
// ------------------------------------------------------------------
__global__ __launch_bounds__(256) void k_log_softmax(float* __restrict__ out, int nrows) {
  int row  = blockIdx.x * 8 + (threadIdx.x >> 5);
  int lane = threadIdx.x & 31;
  if (row >= nrows) return;
  float* p = out + (size_t)row * NCLASS;
  float v0 = p[lane];
  float v1 = p[lane + 32];
  float mx = fmaxf(v0, v1);
#pragma unroll
  for (int o = 16; o > 0; o >>= 1) mx = fmaxf(mx, __shfl_xor(mx, o, 32));
  float s = __expf(v0 - mx) + __expf(v1 - mx);
#pragma unroll
  for (int o = 16; o > 0; o >>= 1) s += __shfl_xor(s, o, 32);
  float lse = mx + __logf(s);
  p[lane]      = v0 - lse;
  p[lane + 32] = v1 - lse;
}

// ------------------------------------------------------------------
extern "C" void kernel_launch(void* const* d_in, const int* in_sizes, int n_in,
                              void* d_out, int out_size, void* d_ws, size_t ws_size,
                              hipStream_t stream) {
  const float*     x    = (const float*)d_in[0];
  const long long* ei   = (const long long*)d_in[1];
  const float*     W1   = (const float*)d_in[2];
  const float*     b1   = (const float*)d_in[3];
  const float*     W2   = (const float*)d_in[4];
  const float*     b2   = (const float*)d_in[5];
  const float*     temp = (const float*)d_in[6];

  const int N = in_sizes[0] / NFEAT;    // 100000
  const int E = in_sizes[1] / 2;        // 3200000
  const long long* src = ei;            // edge_index[0] (row / source)
  const long long* dst = ei + E;        // edge_index[1] (col / target)

  // workspace layout (floats): dinv[N] | norm[E] | h_a[N*64] | h_b[N*64] | Wt1,Wt2 (bf16)
  float* dinv = (float*)d_ws;
  float* nrm  = dinv + N;
  float* h_a  = nrm + E;
  float* h_b  = h_a + (size_t)N * NCLASS;
  size_t wtOff = (size_t)((char*)(h_b + (size_t)N * NCLASS) - (char*)d_ws);
  wtOff = (wtOff + 31) & ~(size_t)31;   // 32 B align for v16bf loads
  __bf16* Wt1 = (__bf16*)((char*)d_ws + wtOff);              // [HIDDEN][NFEAT]
  __bf16* Wt2 = Wt1 + (size_t)HIDDEN * NFEAT;                // [NCLASS][HIDDEN]
  float* out  = (float*)d_out;

  const int T   = 256;
  const int NF4 = N * (NCLASS / 4);     // float4 count of an h buffer
  const int gN   = (N + T - 1) / T;
  const int gE   = (E + T - 1) / T;
  const int gNF4 = (NF4 + T - 1) / T;
  const int gEF  = (int)(((long long)E * 16 + T - 1) / T);

  // ---- gcn_norm ----
  k_init_deg<<<gN, T, 0, stream>>>(dinv, N);
  k_edge_deg<<<gE, T, 0, stream>>>(dst, dinv, E);
  k_rsqrt<<<gN, T, 0, stream>>>(dinv, N);
  k_norm<<<gE, T, 0, stream>>>(src, dst, dinv, nrm, E);

  // ---- weight pre-transpose/convert (one-shot, tiny) ----
  k_cvt_wT<<<(NFEAT * HIDDEN + T - 1) / T, T, 0, stream>>>(W1, Wt1, NFEAT, HIDDEN);
  k_cvt_wT<<<(HIDDEN * NCLASS + T - 1) / T, T, 0, stream>>>(W2, Wt2, HIDDEN, NCLASS);

  // ---- fused MLP (TDM staging + bf16 WMMA) ----
  k_mlp<<<N / 32, T, 0, stream>>>(x, Wt1, b1, Wt2, b2, h_a);

  // ---- GPR propagation ----
  k_scale_init<<<gNF4, T, 0, stream>>>((const float4*)h_a, temp, (float4*)out, NF4);
  float* hc = h_a;
  float* hn = h_b;
  for (int k = 0; k < KPROP; ++k) {
    k_self_init<<<gNF4, T, 0, stream>>>((const float4*)hc, dinv, (float4*)hn, NF4);
    k_edge_prop<<<gEF, T, 0, stream>>>(src, dst, nrm, hc, hn, E);
    k_accum<<<gNF4, T, 0, stream>>>((const float4*)hn, temp + (k + 1), (float4*)out, NF4);
    float* t = hc; hc = hn; hn = t;
  }

  // ---- log_softmax (in place on d_out) ----
  k_log_softmax<<<(N + 7) / 8, T, 0, stream>>>(out, N);
}
